// RNNCute_27590869909586
// MI455X (gfx1250) — matrix operand order repaired and checked by reading, not compiled, not run on hardware
//
#include <hip/hip_runtime.h>
#include <hip/hip_bf16.h>
#include <math.h>
#include <cstdint>

// Problem constants (match reference)
#define BB   4
#define SS_T 4096
#define DIN  2048
#define NH   32
#define SSD  64
#define DOUT 2048
// H*SS == 2048 == DIN == DOUT; K = 2048 for both GEMMs

typedef __attribute__((ext_vector_type(16))) __bf16 v16bf;
typedef __attribute__((ext_vector_type(8)))  __bf16 v8bf;
typedef __attribute__((ext_vector_type(8)))  float  v8f;

constexpr int BM = 128, BN = 128, BK = 64, LDT = BK + 8;  // pad 8 halfs (16 B)
constexpr uint32_t TILE_BYTES = (uint32_t)BM * LDT * 2;   // 18432 B per buffer

// 32-bit LDS byte offset of a __shared__ object (addrspace(3) pointers are 32-bit)
__device__ __forceinline__ uint32_t lds_off_u32(const void* p) {
  return (uint32_t)(uintptr_t)(__attribute__((address_space(3))) const void*)p;
}

// Async DMA: global -> LDS, 16 B per lane, tracked by ASYNCcnt (gfx1250)
__device__ __forceinline__ void async_b128(uint32_t lds_off, uint32_t goff,
                                           const void* sbase) {
  asm volatile("global_load_async_to_lds_b128 %0, %1, %2"
               :: "v"(lds_off), "v"(goff), "s"((uint64_t)(uintptr_t)sbase)
               : "memory");
}
__device__ __forceinline__ void wait_async0() {
  asm volatile("s_wait_asynccnt 0x0" ::: "memory");
}

// Issue one 128x64 bf16 tile (4 async b128 per thread, 256 threads)
__device__ __forceinline__ void stage_tile(const __bf16* __restrict__ gbase,
                                           int row0, int K, int k0,
                                           uint32_t lds_base, int tid) {
#pragma unroll
  for (int it = 0; it < 4; ++it) {
    const int idx = (tid + it * 256) * 8;        // 8 bf16 = 16 B per chunk
    const int r = idx >> 6, c = idx & 63;
    const uint32_t goff = (uint32_t)(((row0 + r) * K + k0 + c) * 2);
    const uint32_t loff = lds_base + (uint32_t)(r * (LDT * 2) + c * 2);
    async_b128(loff, goff, gbase);
  }
}

// ---------------------------------------------------------------------------
// f32 -> bf16 bulk conversion (8 elements / thread, v_cvt_pk_bf16_f32 path)
// ---------------------------------------------------------------------------
__global__ __launch_bounds__(256)
void cvt_bf16(const float* __restrict__ s, __bf16* __restrict__ d, int n8) {
  const int i = blockIdx.x * blockDim.x + threadIdx.x;
  if (i < n8) {
    const float4 a = ((const float4*)s)[2 * i];
    const float4 b = ((const float4*)s)[2 * i + 1];
    v8bf v;
    v[0] = (__bf16)a.x; v[1] = (__bf16)a.y; v[2] = (__bf16)a.z; v[3] = (__bf16)a.w;
    v[4] = (__bf16)b.x; v[5] = (__bf16)b.y; v[6] = (__bf16)b.z; v[7] = (__bf16)b.w;
    *(v8bf*)(d + 8 * (size_t)i) = v;
  }
}

// ---------------------------------------------------------------------------
// bf16 WMMA GEMM:  C[M,N] = A[M,K] * W[N,K]^T (+ bias[n]),  f32 accumulate.
// 256 threads = 8 waves; 128x128 tile; BK=64; double-buffered LDS filled by
// global_load_async_to_lds_b128 (ASYNCcnt) so DMA overlaps WMMA compute.
// Wave grid 4(M) x 2(N); wave tile 32x64 -> 2x4 of 16x16 accumulators.
// ---------------------------------------------------------------------------
template <bool HAS_BIAS>
__global__ __launch_bounds__(256)
void gemm_bf16_wmma(const __bf16* __restrict__ A,
                    const __bf16* __restrict__ Bw,
                    const float* __restrict__ bias,
                    float* __restrict__ C,
                    int M, int N, int K) {
  __shared__ __align__(16) __bf16 As[2][BM][LDT];
  __shared__ __align__(16) __bf16 Bs[2][BN][LDT];

  const int tid  = threadIdx.x;
  const int lane = tid & 31;
  const int wv   = tid >> 5;
  const int wm   = wv & 3;   // 0..3 : 32-row chunk
  const int wn   = wv >> 2;  // 0..1 : 64-col chunk
  const int lr   = lane & 15;
  const int lh   = lane >> 4;

  const int n0 = blockIdx.x * BN;
  const int m0 = blockIdx.y * BM;

  const uint32_t ldsA = lds_off_u32(&As[0][0][0]);
  const uint32_t ldsB = lds_off_u32(&Bs[0][0][0]);

  v8f acc[2][4];
  const v8f vzero = {};
#pragma unroll
  for (int mt = 0; mt < 2; ++mt)
#pragma unroll
    for (int nt = 0; nt < 4; ++nt) acc[mt][nt] = vzero;

  // prologue: fill buffer 0
  stage_tile(A,  m0, K, 0, ldsA, tid);
  stage_tile(Bw, n0, K, 0, ldsB, tid);

  const int T = K / BK;
  for (int i = 0; i < T; ++i) {
    const int cur = i & 1;

    wait_async0();        // this wave's in-flight copies (current buffer) done
    __syncthreads();      // all waves loaded cur & finished computing prev

    if (i + 1 < T) {      // DMA next tile into the other buffer during compute
      const int nxt = cur ^ 1;
      stage_tile(A,  m0, K, (i + 1) * BK, ldsA + (uint32_t)nxt * TILE_BYTES, tid);
      stage_tile(Bw, n0, K, (i + 1) * BK, ldsB + (uint32_t)nxt * TILE_BYTES, tid);
    }

    const __bf16(*Ac)[LDT] = As[cur];
    const __bf16(*Bc)[LDT] = Bs[cur];

#pragma unroll
    for (int kk = 0; kk < BK; kk += 32) {
      v16bf af[2], bfm[4];
#pragma unroll
      for (int mt = 0; mt < 2; ++mt) {
        // A 16x32: lane lr = row M; halfs 0..7 = K[kk+lh*8..], 8..15 = +16
        const __bf16* pa = &Ac[wm * 32 + mt * 16 + lr][kk + lh * 8];
        v8bf lo = *(const v8bf*)pa;
        v8bf hi = *(const v8bf*)(pa + 16);
        v16bf a;
#pragma unroll
        for (int q = 0; q < 8; ++q) { a[q] = lo[q]; a[q + 8] = hi[q]; }
        af[mt] = a;
      }
#pragma unroll
      for (int nt = 0; nt < 4; ++nt) {
        // B 32x16: lane lr = col N; halfs 0..7 = K[kk+lh*8..], 8..15 = +16
        const __bf16* pb = &Bc[wn * 64 + nt * 16 + lr][kk + lh * 8];
        v8bf b0 = *(const v8bf*)pb;
        v8bf b1 = *(const v8bf*)(pb + 16);
        v16bf b;
#pragma unroll
        for (int q = 0; q < 8; ++q) { b[q] = b0[q]; b[q + 8] = b1[q]; }
        bfm[nt] = b;
      }
#pragma unroll
      for (int mt = 0; mt < 2; ++mt)
#pragma unroll
        for (int nt = 0; nt < 4; ++nt)
          acc[mt][nt] = __builtin_amdgcn_wmma_f32_16x16x32_bf16(
              false, af[mt], false, bfm[nt], (short)0, acc[mt][nt], false, false);
    }
  }

  // epilogue: C/D 16x16 f32 layout: VGPR r -> row r + 8*lh, col lr
#pragma unroll
  for (int mt = 0; mt < 2; ++mt) {
#pragma unroll
    for (int nt = 0; nt < 4; ++nt) {
      const int col  = n0 + wn * 64 + nt * 16 + lr;
      const int rowb = m0 + wm * 32 + mt * 16 + lh * 8;
      float bv = 0.0f;
      if constexpr (HAS_BIAS) bv = bias[col];
#pragma unroll
      for (int r = 0; r < 8; ++r)
        C[(size_t)(rowb + r) * N + col] = acc[mt][nt][r] + bv;
    }
  }
}

// ---------------------------------------------------------------------------
// Sequential scan: one block per (b,h); 64 threads; thread i owns output row i
// with SW[h][i][:] register-resident. y_{t-1} broadcast through LDS.
// Emits y directly in bf16 for GEMM2's A operand.
// ---------------------------------------------------------------------------
__global__ __launch_bounds__(64)
void rnn_scan(const float* __restrict__ hx,
              const float* __restrict__ sw,
              const float* __restrict__ init_state,
              void* __restrict__ ybuf_raw) {
  __bf16* ybuf = (__bf16*)ybuf_raw;
  const int b = blockIdx.x >> 5;   // / NH
  const int h = blockIdx.x & 31;   // % NH
  const int i = threadIdx.x;       // 0..63

  float w[SSD];
#pragma unroll
  for (int j = 0; j < SSD; ++j)
    w[j] = sw[((size_t)(h * SSD + i)) * SSD + j];

  __shared__ float yl[SSD];
  yl[i] = init_state[((size_t)b * NH + h) * SSD + i];
  __syncthreads();

  const float* hxp = hx   + ((size_t)b * SS_T) * 2048 + h * SSD + i;
  __bf16*      yp  = ybuf + ((size_t)b * SS_T) * 2048 + h * SSD + i;

  float hv = hxp[0];                          // bias already folded into hx
  for (int t = 0; t < SS_T; ++t) {
    const float hn = (t + 1 < SS_T) ? hxp[(size_t)(t + 1) * 2048] : 0.0f;
    float acc = hv;
#pragma unroll
    for (int j = 0; j < SSD; j += 4) {
      const float4 y4 = *(const float4*)&yl[j];
      acc += w[j] * y4.x + w[j + 1] * y4.y + w[j + 2] * y4.z + w[j + 3] * y4.w;
    }
    const float yv = tanhf(acc);
    __syncthreads();                          // everyone done reading yl
    yl[i] = yv;
    yp[(size_t)t * 2048] = (__bf16)yv;
    __syncthreads();                          // yl ready for next step
    hv = hn;
  }
}

// ---------------------------------------------------------------------------
extern "C" void kernel_launch(void* const* d_in, const int* in_sizes, int n_in,
                              void* d_out, int out_size, void* d_ws, size_t ws_size,
                              hipStream_t stream) {
  const float* x    = (const float*)d_in[0];  // [B,S,DIN]
  const float* init = (const float*)d_in[1];  // [B,H,SS]
  const float* Win  = (const float*)d_in[2];  // [H*SS, DIN]
  const float* SW   = (const float*)d_in[3];  // [H,SS,SS]
  const float* bias = (const float*)d_in[4];  // [H,SS]
  const float* Wout = (const float*)d_in[5];  // [DOUT, H*SS]
  float* out = (float*)d_out;                 // [B,S,DOUT] f32

  const int M = BB * SS_T;                    // 16384
  char* ws = (char*)d_ws;
  float*  hx   = (float*)ws;                                          // 128 MiB
  __bf16* ybuf = (__bf16*)(ws + (size_t)M * 2048 * 4);                //  64 MiB
  __bf16* xbf  = (__bf16*)(ws + (size_t)M * 2048 * 6);                //  64 MiB
  __bf16* winb = (__bf16*)(ws + (size_t)M * 2048 * 8);                //   8 MiB
  __bf16* woutb= (__bf16*)(ws + (size_t)M * 2048 * 8 + (size_t)2048 * 2048 * 2);

  // 1) bulk f32 -> bf16 conversions
  {
    const int n8x = M * 2048 / 8;             // 4194304
    cvt_bf16<<<n8x / 256, 256, 0, stream>>>(x, xbf, n8x);
    const int n8w = 2048 * 2048 / 8;          // 524288
    cvt_bf16<<<n8w / 256, 256, 0, stream>>>(Win,  winb,  n8w);
    cvt_bf16<<<n8w / 256, 256, 0, stream>>>(Wout, woutb, n8w);
  }

  dim3 grid(2048 / BN, M / BM);               // (16, 128)

  // 2) GEMM1: hx = x @ Win^T + bias
  gemm_bf16_wmma<true><<<grid, 256, 0, stream>>>(xbf, winb, bias, hx, M, 2048, DIN);

  // 3) Scan: y_t = tanh(SW @ y_{t-1} + hx_t), y emitted as bf16
  rnn_scan<<<BB * NH, 64, 0, stream>>>(hx, SW, init, ybuf);

  // 4) GEMM2: out = y @ Wout^T
  gemm_bf16_wmma<false><<<grid, 256, 0, stream>>>(ybuf, woutb, nullptr, out, M, DOUT, 2048);
}
